// Memory_70497593197117
// MI455X (gfx1250) — compile-verified, matrix operand order checked by reading, not run on hardware
//
#include <hip/hip_runtime.h>
#include <hip/hip_bf16.h>

typedef _Float16 v8h  __attribute__((ext_vector_type(8)));
typedef _Float16 v16h __attribute__((ext_vector_type(16)));
typedef float    v8f  __attribute__((ext_vector_type(8)));

#define N_PIX   32768      // 8 * 64 * 64
#define DCH     512
#define M_KEYS  2000
#define NTILE   128        // rows per GEMM block (8 waves x 16 rows)
#define LDB     520        // LDS row stride in halves (512 + 8 pad -> bank stride 4)

#if __has_builtin(__builtin_amdgcn_tensor_load_to_lds) && \
    __has_builtin(__builtin_amdgcn_s_wait_tensorcnt)
#define USE_TDM 1
#endif

#define SHUF16(a, b) __builtin_shufflevector((a), (b), 0, 1, 2, 3, 4, 5, 6, 7, \
                                             8, 9, 10, 11, 12, 13, 14, 15)

// ---------------------------------------------------------------------------
// Kernel 1: keys (2000x512 f32) -> f16 row-major + bias[m] = -0.5*|k_m|^2
// ---------------------------------------------------------------------------
__global__ void pack_keys_kernel(const float* __restrict__ keys,
                                 _Float16* __restrict__ kh,
                                 float* __restrict__ bias) {
    const int m    = blockIdx.x * 8 + (threadIdx.x >> 5);
    const int lane = threadIdx.x & 31;
    if (m >= M_KEYS) return;
    const float* kr = keys + (size_t)m * DCH;
    float sum = 0.f;
    #pragma unroll
    for (int i = 0; i < DCH / 32; ++i) {
        float v = kr[lane + 32 * i];
        kh[(size_t)m * DCH + lane + 32 * i] = (_Float16)v;
        sum += v * v;
    }
    #pragma unroll
    for (int off = 16; off >= 1; off >>= 1) sum += __shfl_xor(sum, off, 32);
    if (lane == 0) bias[m] = -0.5f * sum;
}

// ---------------------------------------------------------------------------
// Kernel 2: per-pixel L2 norm over C (BCHW), write normalized q as f16
// row-major (N x 512) via LDS transpose; store 1/norm.
// ---------------------------------------------------------------------------
__global__ void normalize_pack_kernel(const float* __restrict__ query,
                                      _Float16* __restrict__ qh,
                                      float* __restrict__ inv_norm) {
    __shared__ _Float16 lds[256 * 72];
    const int tid  = threadIdx.x;
    const int pix0 = blockIdx.x * 256;
    const int pix  = pix0 + tid;
    const int b    = pix >> 12;            // H*W = 4096
    const int hw   = pix & 4095;
    const size_t base = ((size_t)b * DCH) << 12;

    float sum = 0.f;
    for (int d = 0; d < DCH; ++d) {
        float v = query[base + ((size_t)d << 12) + hw];
        sum += v * v;
    }
    const float inv = 1.0f / fmaxf(sqrtf(sum), 1e-12f);
    inv_norm[pix] = inv;

    for (int d0 = 0; d0 < DCH; d0 += 64) {
        for (int dd = 0; dd < 64; ++dd) {
            float v = query[base + ((size_t)(d0 + dd) << 12) + hw] * inv;
            lds[tid * 72 + dd] = (_Float16)v;
        }
        __syncthreads();
        #pragma unroll
        for (int g = 0; g < 8; ++g) {
            int pl = g * 32 + (tid >> 3);
            int ch = tid & 7;
            *(uint4*)(qh + (size_t)(pix0 + pl) * DCH + d0 + ch * 8) =
                *(const uint4*)(lds + pl * 72 + ch * 8);
        }
        __syncthreads();
    }
}

// ---------------------------------------------------------------------------
// TDM staging: DMA one 16-row x 1024B key tile into LDS, padding each 1024B
// row with 16B (pad_interval=256 DW, pad_amount=4 DW) to match LDB=520.
// ---------------------------------------------------------------------------
#ifdef USE_TDM
typedef unsigned int u32x4 __attribute__((ext_vector_type(4)));
typedef int          i32x4 __attribute__((ext_vector_type(4)));
typedef int          i32x8 __attribute__((ext_vector_type(8)));

__device__ __forceinline__ void tdm_stage(_Float16* ldsDst, const _Float16* gsrc) {
    const unsigned long long ga = (unsigned long long)(uintptr_t)gsrc;
    const unsigned lds_addr = (unsigned)(uintptr_t)ldsDst;   // low 32b = LDS offset
    u32x4 g0;
    g0.x = 1u;                                               // count=1, user mode
    g0.y = lds_addr;                                         // lds_addr[31:0]
    g0.z = (unsigned)(ga & 0xFFFFFFFFu);                     // global_addr[31:0]
    g0.w = (unsigned)((ga >> 32) & 0x1FFFFFFu) | (2u << 30); // addr[56:32] | type=2
    i32x8 g1;
    g1[0] = (1 << 16)      // data_size = 2 bytes
          | (1 << 20)      // pad_enable
          | (7 << 22)      // pad_interval: 256 DWORDs (one 1024B row)
          | (3 << 25);     // pad_amount: 4 DWORDs (16B)
    g1[1] = (int)(512u << 16);   // tensor_dim0[15:0]=512 (atomic_barrier_addr=0)
    g1[2] = (int)(16u << 16);    // tensor_dim0[31:16]=0 | tensor_dim1[15:0]=16
    g1[3] = (int)(512u << 16);   // tensor_dim1[31:16]=0 | tile_dim0=512
    g1[4] = 16;                  // tile_dim1=16 | tile_dim2=0
    g1[5] = 512;                 // tensor_dim0_stride[31:0]=512 elements
    g1[6] = 0;                   // stride0[47:32]=0 | stride1[15:0]=0
    g1[7] = 0;
    const i32x4 z4 = {0, 0, 0, 0};
    const i32x8 z8 = {0, 0, 0, 0, 0, 0, 0, 0};
    __builtin_amdgcn_tensor_load_to_lds(g0, g1, z4, z4, z8, 0);
}
#endif

__device__ __forceinline__ void stage_tile_coop(_Float16* dst, const _Float16* src,
                                                int tid) {
    for (int idx = tid; idx < 16 * 64; idx += 256) {
        int r = idx >> 6, c = (idx & 63) << 3;
        *(uint4*)(dst + r * LDB + c) = *(const uint4*)(src + (size_t)r * DCH + c);
    }
}

// ---------------------------------------------------------------------------
// Kernel 3: WMMA GEMM fused with biased argmax.
// A (16 rows x 512 f16 per wave) resident in 128 VGPRs; B streams through a
// double-buffered LDS tile filled by the Tensor Data Mover. Split-K dual
// accumulators + depth-1 B prefetch keep the WMMA pipe fed.
// ---------------------------------------------------------------------------
__global__ void __launch_bounds__(256)
gemm_argmax_kernel(const _Float16* __restrict__ qh,
                   const _Float16* __restrict__ kh,
                   const float* __restrict__ bias,
                   int* __restrict__ nearest) {
    __shared__ _Float16 kbuf[2][16 * LDB];    // 2 x 16.6 KB

    const int tid      = threadIdx.x;
    const int lane     = tid & 31;
    const int wave     = tid >> 5;
    const int l15      = lane & 15;
    const int halfSel  = lane >> 4;           // 0: K{0..7,16..23} 1: K{8..15,24..31}
    const int kbase    = halfSel ? 8 : 0;
    const int rowBlock = blockIdx.x * NTILE;
    const int waveRow  = wave * 16;

    // A operand: this wave's 16 rows, full K, resident in registers (read once)
    v8h areg[32];
    {
        const _Float16* aptr = qh + (size_t)(rowBlock + waveRow + l15) * DCH + kbase;
        #pragma unroll
        for (int i = 0; i < 16; ++i) {
            areg[2 * i]     = *(const v8h*)(aptr + i * 32);
            areg[2 * i + 1] = *(const v8h*)(aptr + i * 32 + 16);
        }
    }

    float best[8];
    int   bidx[8];
    #pragma unroll
    for (int j = 0; j < 8; ++j) { best[j] = -3.0e38f; bidx[j] = 0; }

    // stage tile 0
#ifdef USE_TDM
    if (wave == 0) {
        tdm_stage(&kbuf[0][0], kh);
        __builtin_amdgcn_s_wait_tensorcnt(0);
    }
#else
    stage_tile_coop(&kbuf[0][0], kh, tid);
#endif
    __syncthreads();

    for (int t = 0; t < M_KEYS / 16; ++t) {
        const int cur = t & 1;
        // prefetch next tile into the other buffer while computing this one
#ifdef USE_TDM
        if (wave == 0 && t + 1 < M_KEYS / 16)
            tdm_stage(&kbuf[1 - cur][0], kh + (size_t)(t + 1) * 16 * DCH);
#else
        if (t + 1 < M_KEYS / 16)
            stage_tile_coop(&kbuf[1 - cur][0], kh + (size_t)(t + 1) * 16 * DCH, tid);
#endif

        const int   ncol = t * 16 + l15;
        const float bl   = bias[ncol];          // issued ahead of the WMMA chain

        // chain0: K[0..255] -> acc0 ; chain1: K[256..511] -> acc1
        v8f acc0 = {0.f, 0.f, 0.f, 0.f, 0.f, 0.f, 0.f, 0.f};
        v8f acc1 = {0.f, 0.f, 0.f, 0.f, 0.f, 0.f, 0.f, 0.f};
        const _Float16* brow = &kbuf[cur][l15 * LDB + kbase];

        v8h pb0 = *(const v8h*)(brow);
        v8h pb1 = *(const v8h*)(brow + 16);
        v8h qb0 = *(const v8h*)(brow + 256);
        v8h qb1 = *(const v8h*)(brow + 256 + 16);
        #pragma unroll
        for (int i = 0; i < 7; ++i) {
            // prefetch step i+1 for both chains before using step i
            v8h nb0 = *(const v8h*)(brow + (i + 1) * 32);
            v8h nb1 = *(const v8h*)(brow + (i + 1) * 32 + 16);
            v8h nq0 = *(const v8h*)(brow + 256 + (i + 1) * 32);
            v8h nq1 = *(const v8h*)(brow + 256 + (i + 1) * 32 + 16);
            acc0 = __builtin_amdgcn_wmma_f32_16x16x32_f16(
                       false, SHUF16(areg[2 * i], areg[2 * i + 1]),
                       false, SHUF16(pb0, pb1), (short)0, acc0, false, false);
            acc1 = __builtin_amdgcn_wmma_f32_16x16x32_f16(
                       false, SHUF16(areg[2 * (i + 8)], areg[2 * (i + 8) + 1]),
                       false, SHUF16(qb0, qb1), (short)0, acc1, false, false);
            pb0 = nb0; pb1 = nb1; qb0 = nq0; qb1 = nq1;
        }
        acc0 = __builtin_amdgcn_wmma_f32_16x16x32_f16(
                   false, SHUF16(areg[14], areg[15]),
                   false, SHUF16(pb0, pb1), (short)0, acc0, false, false);
        acc1 = __builtin_amdgcn_wmma_f32_16x16x32_f16(
                   false, SHUF16(areg[30], areg[31]),
                   false, SHUF16(qb0, qb1), (short)0, acc1, false, false);

        #pragma unroll
        for (int j = 0; j < 8; ++j) {
            float s = acc0[j] + acc1[j] + bl;
            if (s > best[j]) { best[j] = s; bidx[j] = ncol; }
        }

#ifdef USE_TDM
        if (wave == 0 && t + 1 < M_KEYS / 16)
            __builtin_amdgcn_s_wait_tensorcnt(0);   // TDM(t+1) overlapped compute(t)
#endif
        __syncthreads();
    }

    // argmax across the 16 column-lanes of each half-wave
    #pragma unroll
    for (int off = 8; off >= 1; off >>= 1) {
        #pragma unroll
        for (int j = 0; j < 8; ++j) {
            float ov = __shfl_xor(best[j], off, 32);
            int   oi = __shfl_xor(bidx[j], off, 32);
            if (ov > best[j] || (ov == best[j] && oi < bidx[j])) {
                best[j] = ov; bidx[j] = oi;
            }
        }
    }
    if (l15 == 0) {
        #pragma unroll
        for (int j = 0; j < 8; ++j)
            nearest[rowBlock + waveRow + j + 8 * halfSel] = bidx[j];
    }
}

// ---------------------------------------------------------------------------
// Kernel 4: heatmap = sum_d (q_norm - keys[nearest])^4 in full FP32.
// ---------------------------------------------------------------------------
__global__ void heatmap_kernel(const float* __restrict__ query,
                               const float* __restrict__ keys,
                               const float* __restrict__ inv_norm,
                               const int* __restrict__ nearest,
                               float* __restrict__ out) {
    const int pix = blockIdx.x * 256 + threadIdx.x;
    const int b   = pix >> 12;
    const int hw  = pix & 4095;
    const size_t base = ((size_t)b * DCH) << 12;
    const float inv = inv_norm[pix];
    const float* kr = keys + (size_t)nearest[pix] * DCH;
    float acc = 0.f;
    for (int d = 0; d < DCH; ++d) {
        float qv = query[base + ((size_t)d << 12) + hw] * inv;
        float df = qv - kr[d];
        float d2 = df * df;
        acc += d2 * d2;
    }
    out[pix] = acc;
}

// ---------------------------------------------------------------------------
extern "C" void kernel_launch(void* const* d_in, const int* in_sizes, int n_in,
                              void* d_out, int out_size, void* d_ws, size_t ws_size,
                              hipStream_t stream) {
    const float* query = (const float*)d_in[0];   // (8,512,64,64) f32
    const float* keys  = (const float*)d_in[1];   // (2000,512)   f32
    float* out = (float*)d_out;                   // (8,64,64,1)  f32

    char* ws = (char*)d_ws;
    size_t off = 0;
    _Float16* qh = (_Float16*)(ws + off); off += (size_t)N_PIX * DCH * 2;   // 32 MB
    _Float16* kh = (_Float16*)(ws + off); off += (size_t)M_KEYS * DCH * 2;  // 2 MB
    float* bias     = (float*)(ws + off); off += ((size_t)M_KEYS * 4 + 255) & ~255ull;
    float* inv_norm = (float*)(ws + off); off += (size_t)N_PIX * 4;
    int*   nearest  = (int*)  (ws + off); off += (size_t)N_PIX * 4;
    (void)ws_size; (void)in_sizes; (void)n_in; (void)out_size;

    pack_keys_kernel<<<M_KEYS / 8, 256, 0, stream>>>(keys, kh, bias);
    normalize_pack_kernel<<<N_PIX / 256, 256, 0, stream>>>(query, qh, inv_norm);
    gemm_argmax_kernel<<<N_PIX / NTILE, 256, 0, stream>>>(qh, kh, bias, nearest);
    heatmap_kernel<<<N_PIX / 256, 256, 0, stream>>>(query, keys, inv_norm, nearest, out);
}
